// MoE_5523327943231
// MI455X (gfx1250) — matrix operand order repaired and checked by reading, not compiled, not run on hardware
//
#include <hip/hip_runtime.h>
#include <hip/hip_bf16.h>

// ---------------- types ----------------
typedef __bf16 bf16;
typedef __attribute__((ext_vector_type(16))) __bf16 v16bf;
typedef __attribute__((ext_vector_type(8)))  __bf16 v8bf;
typedef __attribute__((ext_vector_type(8)))  float  v8f;
typedef int v4i __attribute__((vector_size(16)));   // matches builtin param type

#if __has_builtin(__builtin_amdgcn_global_load_async_to_lds_b128)
#define USE_ASYNC_LDS 1
#else
#define USE_ASYNC_LDS 0
#endif

// ---------------- problem constants ----------------
#define T_TOK 2048   // B*S
#define DDIM  1024
#define IDIM  704
#define SIDIM 1408
#define NEXP  64
#define TOPK  6
#define CAPE  512

static __device__ __forceinline__ bf16 f2bf(float f) {
    union { float f; unsigned u; } un; un.f = f;
    unsigned r = un.u + 0x7FFFu + ((un.u >> 16) & 1u);   // round-to-nearest-even
    unsigned short h = (unsigned short)(r >> 16);
    return __builtin_bit_cast(bf16, h);
}

// silu(g) = g * sigmoid(g); fast v_rcp_f32 path (bf16-storage precision is fine)
static __device__ __forceinline__ float silu_fast(float g) {
    return g * __builtin_amdgcn_rcpf(1.f + __expf(-g));
}

// Build a wave32 16x32 bf16 A-matrix fragment (contiguous k order, same as the B loads) from a row-major row.
// lanes 0-15 (hi=0): K = 0..15 ; lanes 16-31 (hi=1): K = 16..31
static __device__ __forceinline__ v16bf load_a_frag(const bf16* __restrict__ row,
                                                    int k0, int hi) {
    v8bf lo = *(const v8bf*)(row + k0 + hi * 16);
    v8bf h8 = *(const v8bf*)(row + k0 + hi * 16 + 8);
    v16bf a;
    #pragma unroll
    for (int j = 0; j < 8; ++j) { a[j] = lo[j]; a[8 + j] = h8[j]; }
    return a;
}

// ---------------- utility kernels ----------------
__global__ void zero_f32_kernel(float* __restrict__ p, int n) {
    int i = blockIdx.x * blockDim.x + threadIdx.x;
    if (i < n) p[i] = 0.0f;
}

__global__ void cvt_bf16_kernel(const float* __restrict__ s, bf16* __restrict__ d, int n) {
    int i = blockIdx.x * blockDim.x + threadIdx.x;
    if (i < n) d[i] = f2bf(s[i]);
}

// ---------------- gate: softmax + top-k + dispatch ----------------
// one wave (32 lanes) per token; lane handles experts (lane) and (lane+32)
__global__ void gate_topk_kernel(const float* __restrict__ x,
                                 const float* __restrict__ gw,
                                 const float* __restrict__ gb,
                                 int*   __restrict__ counts,
                                 int*   __restrict__ toklist,
                                 float* __restrict__ wlist) {
    const int t    = blockIdx.x * 4 + (threadIdx.x >> 5);
    const int lane = threadIdx.x & 31;
    if (t >= T_TOK) return;

    const float4* xr = (const float4*)(x + (size_t)t * DDIM);
    const int e0 = lane, e1 = lane + 32;
    const float4* g0 = (const float4*)(gw + (size_t)e0 * DDIM);
    const float4* g1 = (const float4*)(gw + (size_t)e1 * DDIM);

    float s0 = 0.f, s1 = 0.f;
    for (int d = 0; d < DDIM / 4; ++d) {
        float4 xv = xr[d];
        float4 a  = g0[d];
        float4 b  = g1[d];
        s0 += xv.x * a.x + xv.y * a.y + xv.z * a.z + xv.w * a.w;
        s1 += xv.x * b.x + xv.y * b.y + xv.z * b.z + xv.w * b.w;
    }

    // softmax over 64 logits (2 per lane)
    float m = fmaxf(s0, s1);
    for (int off = 16; off; off >>= 1) m = fmaxf(m, __shfl_xor(m, off, 32));
    float p0 = __expf(s0 - m), p1 = __expf(s1 - m);
    float sum = p0 + p1;
    for (int off = 16; off; off >>= 1) sum += __shfl_xor(sum, off, 32);
    p0 /= sum; p1 /= sum;

    const float b0 = gb[e0], b1 = gb[e1];
    bool used0 = false, used1 = false;

    for (int k = 0; k < TOPK; ++k) {
        float v0 = used0 ? -1e30f : p0 + b0;
        float v1 = used1 ? -1e30f : p1 + b1;
        float v, p; int id;
        if (v1 > v0) { v = v1; p = p1; id = e1; } else { v = v0; p = p0; id = e0; }
        for (int off = 16; off; off >>= 1) {
            float ov = __shfl_xor(v, off, 32);
            float op = __shfl_xor(p, off, 32);
            int  oid = __shfl_xor(id, off, 32);
            if (ov > v || (ov == v && oid < id)) { v = ov; p = op; id = oid; }
        }
        if (id == e0) used0 = true;
        if (id == e1) used1 = true;
        if (lane == 0) {
            int slot = atomicAdd(&counts[id], 1);
            if (slot < CAPE) {
                toklist[id * CAPE + slot] = t;
                wlist[id * CAPE + slot]   = p;  // ROUTE_SCALE == 1.0
            }
        }
    }
}

// ---------------- fused SwiGLU expert GEMM ----------------
// block = 128 threads (4 waves); one block per (expert=blockIdx.y, 16-token tile=blockIdx.x)
// stage:   x-tile (16 x D bf16) -> LDS via async global->LDS copy (ASYNCcnt)
// phase 1: h[16 x I] = silu(x W1^T) * (x W3^T)  -> LDS (bf16)
// phase 2: y[16 x D] = h W2^T -> weighted atomic scatter-add into out
// Waves process PAIRS of 16-wide output tiles; bounded k-loop unrolling keeps
// A fragments from being mass-hoisted/spilled.
// shared-expert mode: counts/toklist/wlist == nullptr -> identity tokens, weight 1
__global__ void fused_swiglu_kernel(const bf16* __restrict__ xb,
                                    const bf16* __restrict__ w1,
                                    const bf16* __restrict__ w3,
                                    const bf16* __restrict__ w2,
                                    const int*  __restrict__ counts,
                                    const int*  __restrict__ toklist,
                                    const float* __restrict__ wlist,
                                    float* __restrict__ out,
                                    int I) {
    extern __shared__ __align__(16) char smem_raw[];
    bf16*  xS    = (bf16*)smem_raw;                                   // 16 x DDIM
    bf16*  hS    = (bf16*)(smem_raw + (size_t)16 * DDIM * sizeof(bf16)); // 16 x I
    int*   s_tok = (int*)(smem_raw + (size_t)16 * DDIM * sizeof(bf16)
                                   + (size_t)16 * I * sizeof(bf16));
    float* s_w   = (float*)(s_tok + 16);

    const int e  = blockIdx.y;
    const int m0 = blockIdx.x * 16;
    const int cnt = counts ? min(counts[e], CAPE) : (int)(gridDim.x * 16);
    if (m0 >= cnt) return;

    const size_t wstride = (size_t)I * DDIM;
    const bf16* w1e = w1 + (size_t)e * wstride;
    const bf16* w3e = w3 + (size_t)e * wstride;
    const bf16* w2e = w2 + (size_t)e * wstride;

    if (threadIdx.x < 16) {
        int g = m0 + (int)threadIdx.x;
        bool valid = g < cnt;
        int tk = 0; float wv = 0.f;
        if (toklist) {
            if (valid) { tk = toklist[e * CAPE + g]; wv = wlist[e * CAPE + g]; }
        } else {
            if (valid) { tk = g; wv = 1.f; }
        }
        s_tok[threadIdx.x] = tk;
        s_w[threadIdx.x]   = wv;
    }
    __syncthreads();

    // ---- stage the 16 x DDIM x-tile into LDS (gathered token rows) ----
    {
        const int chunks_per_row = (DDIM * (int)sizeof(bf16)) / 16;   // 128
        for (int c = threadIdx.x; c < 16 * chunks_per_row; c += blockDim.x) {
            const int row = c >> 7;
            const int cc  = c & (chunks_per_row - 1);
            const bf16* g = xb + (size_t)s_tok[row] * DDIM + cc * 8;
            bf16*       l = xS + (size_t)row * DDIM + cc * 8;
#if USE_ASYNC_LDS
            __builtin_amdgcn_global_load_async_to_lds_b128((v4i*)g, (v4i*)l, 0, 0);
#else
            *(v8bf*)l = *(const v8bf*)g;
#endif
        }
#if USE_ASYNC_LDS
#if __has_builtin(__builtin_amdgcn_s_wait_asynccnt)
        __builtin_amdgcn_s_wait_asynccnt(0);
#else
        asm volatile("s_wait_asynccnt 0x0" ::: "memory");
#endif
#endif
    }
    __syncthreads();

    const int lane = threadIdx.x & 31;
    const int wave = threadIdx.x >> 5;
    const int r    = lane & 15;   // row / column sub-index within a 16-wide tile
    const int hi   = lane >> 4;   // lane half select

    const bf16* xsrow = xS + (size_t)r * DDIM;

    // ---- phase 1: h = silu(x W1^T) * (x W3^T), i-tiles in pairs ----
    const int nIt2 = I >> 5;                  // number of 32-wide i-tile pairs
    for (int it2 = wave; it2 < nIt2; it2 += 4) {
        const int i0 = it2 << 5;              // covers columns i0 .. i0+31
        const bf16* w1rowA = w1e + (size_t)(i0 + r) * DDIM;
        const bf16* w1rowB = w1e + (size_t)(i0 + 16 + r) * DDIM;
        const bf16* w3rowA = w3e + (size_t)(i0 + r) * DDIM;
        const bf16* w3rowB = w3e + (size_t)(i0 + 16 + r) * DDIM;
        v8f acc1a = {0.f,0.f,0.f,0.f,0.f,0.f,0.f,0.f};
        v8f acc1b = {0.f,0.f,0.f,0.f,0.f,0.f,0.f,0.f};
        v8f acc3a = {0.f,0.f,0.f,0.f,0.f,0.f,0.f,0.f};
        v8f acc3b = {0.f,0.f,0.f,0.f,0.f,0.f,0.f,0.f};
        #pragma unroll 2
        for (int k0 = 0; k0 < DDIM; k0 += 32) {
            __builtin_prefetch(w1rowA + k0 + 512, 0, 1);
            __builtin_prefetch(w3rowA + k0 + 512, 0, 1);
            v16bf a = load_a_frag(xsrow, k0, hi);     // from LDS
            v16bf b;
            b = *(const v16bf*)(w1rowA + k0 + hi * 16);
            acc1a = __builtin_amdgcn_wmma_f32_16x16x32_bf16(false, a, false, b,
                                                            (short)0, acc1a, false, false);
            b = *(const v16bf*)(w3rowA + k0 + hi * 16);
            acc3a = __builtin_amdgcn_wmma_f32_16x16x32_bf16(false, a, false, b,
                                                            (short)0, acc3a, false, false);
            b = *(const v16bf*)(w1rowB + k0 + hi * 16);
            acc1b = __builtin_amdgcn_wmma_f32_16x16x32_bf16(false, a, false, b,
                                                            (short)0, acc1b, false, false);
            b = *(const v16bf*)(w3rowB + k0 + hi * 16);
            acc3b = __builtin_amdgcn_wmma_f32_16x16x32_bf16(false, a, false, b,
                                                            (short)0, acc3b, false, false);
        }
        #pragma unroll
        for (int i = 0; i < 8; ++i) {
            const int mrow = i + 8 * hi;
            hS[(size_t)mrow * I + i0 + r]      = f2bf(silu_fast(acc1a[i]) * acc3a[i]);
            hS[(size_t)mrow * I + i0 + 16 + r] = f2bf(silu_fast(acc1b[i]) * acc3b[i]);
        }
    }
    __syncthreads();

    // ---- phase 2: y = h W2^T, d-tiles in pairs, weighted scatter-add ----
    const bf16* hrow = hS + (size_t)r * I;
    const int nDt2 = DDIM >> 5;               // 32
    for (int dt2 = wave; dt2 < nDt2; dt2 += 4) {
        const int d0 = dt2 << 5;              // covers columns d0 .. d0+31
        const bf16* w2rowA = w2e + (size_t)(d0 + r) * I;
        const bf16* w2rowB = w2e + (size_t)(d0 + 16 + r) * I;
        v8f accA = {0.f,0.f,0.f,0.f,0.f,0.f,0.f,0.f};
        v8f accB = {0.f,0.f,0.f,0.f,0.f,0.f,0.f,0.f};
        #pragma unroll 2
        for (int k0 = 0; k0 < I; k0 += 32) {
            __builtin_prefetch(w2rowA + k0 + 512, 0, 1);
            v16bf a = load_a_frag(hrow, k0, hi);   // from LDS
            v16bf b;
            b = *(const v16bf*)(w2rowA + k0 + hi * 16);
            accA = __builtin_amdgcn_wmma_f32_16x16x32_bf16(false, a, false, b,
                                                           (short)0, accA, false, false);
            b = *(const v16bf*)(w2rowB + k0 + hi * 16);
            accB = __builtin_amdgcn_wmma_f32_16x16x32_bf16(false, a, false, b,
                                                           (short)0, accB, false, false);
        }
        #pragma unroll
        for (int i = 0; i < 8; ++i) {
            const int mrow = i + 8 * hi;
            const float wv = s_w[mrow];
            float* dst = out + (size_t)s_tok[mrow] * DDIM + d0 + r;
            __hip_atomic_fetch_add(dst, accA[i] * wv,
                                   __ATOMIC_RELAXED, __HIP_MEMORY_SCOPE_AGENT);
            __hip_atomic_fetch_add(dst + 16, accB[i] * wv,
                                   __ATOMIC_RELAXED, __HIP_MEMORY_SCOPE_AGENT);
        }
    }
}

// ---------------- host side ----------------
extern "C" void kernel_launch(void* const* d_in, const int* in_sizes, int n_in,
                              void* d_out, int out_size, void* d_ws, size_t ws_size,
                              hipStream_t stream) {
    const float* x   = (const float*)d_in[0];
    const float* gw  = (const float*)d_in[1];
    const float* gb  = (const float*)d_in[2];
    const float* w1  = (const float*)d_in[3];
    const float* w2  = (const float*)d_in[4];
    const float* w3  = (const float*)d_in[5];
    const float* ws1 = (const float*)d_in[6];
    const float* ws2 = (const float*)d_in[7];
    const float* ws3 = (const float*)d_in[8];
    float* out = (float*)d_out;

    char* wsb = (char*)d_ws;
    size_t off = 0;
    auto alloc = [&](size_t bytes) -> char* {
        char* p = wsb + off;
        off += (bytes + 255) & ~(size_t)255;
        return p;
    };
    const size_t nXb  = (size_t)T_TOK * DDIM;
    const size_t nWe  = (size_t)NEXP * IDIM * DDIM;
    const size_t nWs  = (size_t)SIDIM * DDIM;

    bf16* xb   = (bf16*)alloc(nXb * sizeof(bf16));
    bf16* w1b  = (bf16*)alloc(nWe * sizeof(bf16));
    bf16* w3b  = (bf16*)alloc(nWe * sizeof(bf16));
    bf16* w2b  = (bf16*)alloc(nWe * sizeof(bf16));
    bf16* ws1b = (bf16*)alloc(nWs * sizeof(bf16));
    bf16* ws3b = (bf16*)alloc(nWs * sizeof(bf16));
    bf16* ws2b = (bf16*)alloc(nWs * sizeof(bf16));
    int*   counts  = (int*)alloc(NEXP * sizeof(int));
    int*   toklist = (int*)alloc((size_t)NEXP * CAPE * sizeof(int));
    float* wlist   = (float*)alloc((size_t)NEXP * CAPE * sizeof(float));

    // zero output and expert counters
    {
        int n = T_TOK * DDIM;
        zero_f32_kernel<<<(n + 255) / 256, 256, 0, stream>>>(out, n);
        zero_f32_kernel<<<1, 64, 0, stream>>>((float*)counts, NEXP);
    }

    // fp32 -> bf16 conversions (weights streamed once; halves GEMM bandwidth)
    {
        int n;
        n = (int)nXb; cvt_bf16_kernel<<<(n + 255) / 256, 256, 0, stream>>>(x,   xb,   n);
        n = (int)nWe; cvt_bf16_kernel<<<(n + 255) / 256, 256, 0, stream>>>(w1,  w1b,  n);
        n = (int)nWe; cvt_bf16_kernel<<<(n + 255) / 256, 256, 0, stream>>>(w3,  w3b,  n);
        n = (int)nWe; cvt_bf16_kernel<<<(n + 255) / 256, 256, 0, stream>>>(w2,  w2b,  n);
        n = (int)nWs; cvt_bf16_kernel<<<(n + 255) / 256, 256, 0, stream>>>(ws1, ws1b, n);
        n = (int)nWs; cvt_bf16_kernel<<<(n + 255) / 256, 256, 0, stream>>>(ws3, ws3b, n);
        n = (int)nWs; cvt_bf16_kernel<<<(n + 255) / 256, 256, 0, stream>>>(ws2, ws2b, n);
    }

    // gate + dispatch
    gate_topk_kernel<<<T_TOK / 4, 128, 0, stream>>>(x, gw, gb, counts, toklist, wlist);

    // routed experts: grid (CAP/16 tiles, E experts)
    {
        size_t lds = (size_t)16 * DDIM * sizeof(bf16)
                   + (size_t)16 * IDIM * sizeof(bf16)
                   + 16 * sizeof(int) + 16 * sizeof(float);
        fused_swiglu_kernel<<<dim3(CAPE / 16, NEXP), 128, lds, stream>>>(
            xb, w1b, w3b, w2b, counts, toklist, wlist, out, IDIM);
    }

    // shared expert: identity routing, weight 1
    {
        size_t lds = (size_t)16 * DDIM * sizeof(bf16)
                   + (size_t)16 * SIDIM * sizeof(bf16)
                   + 16 * sizeof(int) + 16 * sizeof(float);
        fused_swiglu_kernel<<<dim3(T_TOK / 16, 1), 128, lds, stream>>>(
            xb, ws1b, ws3b, ws2b, nullptr, nullptr, nullptr, out, SIDIM);
    }
}
